// CDGMLinear_69406671503636
// MI455X (gfx1250) — matrix operand order repaired and checked by reading, not compiled, not run on hardware
//
#include <hip/hip_runtime.h>

#define NN     16384
#define DIN    128
#define LAT    64
#define HID    128
#define CSPLIT 4                      // column-range split for wave parallelism
#define CHUNK  (NN / CSPLIT)          // 4096 columns per chunk

typedef __bf16 bf16_t;
typedef bf16_t bf16x16 __attribute__((ext_vector_type(16)));
typedef float  f32x8   __attribute__((ext_vector_type(8)));
typedef unsigned int u32x4 __attribute__((ext_vector_type(4)));

union V16 { bf16x16 v; u32x4 q[2]; };

// WMMA wrapper: reuse hints must be integer constant expressions -> template.
template <bool RA>
__device__ __forceinline__ f32x8 wmma_bf16(bf16x16 a, bf16x16 b, f32x8 c)
{
    return __builtin_amdgcn_wmma_f32_16x16x32_bf16(
        false, a, false, b, (short)0, c, RA, false);
}

// Async copy 16 bytes: global (saddr base + voffset) -> LDS (vdst offset).
// Tracked by ASYNCcnt (CDNA5 ISA).
__device__ __forceinline__ void async_ld16(unsigned lds_off, const void* base, unsigned goff)
{
    asm volatile("global_load_async_to_lds_b128 %0, %1, %2"
                 :: "v"(lds_off), "v"(goff), "s"(base) : "memory");
}
__device__ __forceinline__ void wait_async0()
{
    asm volatile("s_wait_asynccnt 0x0" ::: "memory");
}

// ---------------------------------------------------------------------------
// Projection kernel: per row r of x[N,128]:
//   g[r,:64]   = relu(x@gl_w + gl_b)   (bf16, row major)
//   sq[r]      = sum(g^2)
//   hT[:128,r] = x@gnn_w + gnn_b       (bf16, TRANSPOSED: [128][N])
// hT column-major => fused kernel's B tiles are contiguous 64B runs ->
// straight async DMA into LDS, no in-LDS transpose.
// ---------------------------------------------------------------------------
__global__ __launch_bounds__(128) void proj_kernel(
    const float* __restrict__ x,
    const float* __restrict__ glw, const float* __restrict__ glb,
    const float* __restrict__ gnw, const float* __restrict__ gnb,
    bf16_t* __restrict__ g_out, float* __restrict__ sq_out,
    bf16_t* __restrict__ hT_out)
{
    __shared__ float xrow[DIN];
    __shared__ float gsq[LAT];
    const int r = blockIdx.x;
    const int t = threadIdx.x;

    xrow[t] = x[(size_t)r * DIN + t];
    __syncthreads();

    if (t < LAT) {
        float acc = glb[t];
        #pragma unroll 8
        for (int k = 0; k < DIN; ++k)
            acc = fmaf(xrow[k], glw[k * LAT + t], acc);
        acc = fmaxf(acc, 0.0f);
        g_out[(size_t)r * LAT + t] = (bf16_t)acc;
        gsq[t] = acc * acc;
    }

    float acc = gnb[t];
    #pragma unroll 8
    for (int k = 0; k < DIN; ++k)
        acc = fmaf(xrow[k], gnw[k * HID + t], acc);
    hT_out[(size_t)t * NN + r] = (bf16_t)acc;   // transposed store

    __syncthreads();
    if (t == 0) {
        float s = 0.0f;
        #pragma unroll 8
        for (int i = 0; i < LAT; ++i) s += gsq[i];
        sq_out[r] = s;
    }
}

// ---------------------------------------------------------------------------
// Fused adjacency + aggregation, PARTIAL over a column chunk (exact; the
// sigmoid aggregation and its row-sum are both linear in j):
//   pacc[cs,i,:] = sum_{j in chunk cs} sigmoid(2t*g_i.g_j - t*(sq_i+sq_j)+th) * h[j,:]
//   pden[cs,i]   = sum_{j in chunk cs} sigmoid(...)
// blockIdx = {row-group 0..255} x {chunk 0..3} -> 1024 blocks x 4 wave32.
// Each wave: 16-row tile, j-loop step 32, double-buffered async LDS DMA,
// 12 x v_wmma_f32_16x16x32_bf16 per iteration with A-reuse hints.
// ---------------------------------------------------------------------------
__global__ __launch_bounds__(128) void fused_kernel(
    const bf16_t* __restrict__ g, const float* __restrict__ sq,
    const bf16_t* __restrict__ hT,
    const float* __restrict__ temp_p, const float* __restrict__ theta_p,
    float* __restrict__ pacc, float* __restrict__ pden)
{
    __shared__ __align__(16) bf16_t lds_g[2][32 * LAT];    // 2 x 4 KB, row major
    __shared__ __align__(16) bf16_t lds_hT[2][HID * 32];   // 2 x 8 KB, [n][k]
    __shared__ __align__(16) bf16_t lds_P[4][16 * 32];     // per-wave P staging

    const int t    = threadIdx.x;
    const int lane = t & 31;
    const int wv   = t >> 5;
    const int half = lane >> 4;     // 0: lanes 0-15, 1: lanes 16-31
    const int l15  = lane & 15;
    const int rb   = blockIdx.x & 255;      // row group
    const int cs   = blockIdx.x >> 8;       // column chunk
    const int ibase = rb * 64 + wv * 16;
    const int jb = cs * CHUNK, je = jb + CHUNK;

    const float tt = 1.0f + *temp_p;
    const float th = 5.0f + *theta_p;

    // A-operands for g_i (16x64 -> two 16x32 A tiles).
    // A layout (16-bit 16x32): lane half h holds K k0..k0+15 (v0-7),
    // k0 = h*16 (matches B contiguous order).
    bf16x16 Ag[2];
    {
        const int r  = ibase + l15;
        const int k0 = half * 16;
        #pragma unroll
        for (int ks = 0; ks < 2; ++ks) {
            V16 a;
            const bf16_t* src = g + (size_t)r * LAT + ks * 32 + k0;
            a.q[0] = *(const u32x4*)(src);
            a.q[1] = *(const u32x4*)(src + 8);
            Ag[ks] = a.v;
        }
    }
    // sq_i per C/D row M = v + half*8
    float sqi[8];
    #pragma unroll
    for (int v = 0; v < 8; ++v) sqi[v] = sq[ibase + v + half * 8];

    const f32x8 zero8 = {0.f, 0.f, 0.f, 0.f, 0.f, 0.f, 0.f, 0.f};
    f32x8 acc[8];
    #pragma unroll
    for (int n = 0; n < 8; ++n) acc[n] = zero8;
    float dpart[8] = {0.f, 0.f, 0.f, 0.f, 0.f, 0.f, 0.f, 0.f};

    // --- async DMA of one (g, hT) tile pair into LDS buffer `buf` -----------
    auto issue_tile = [&](int buf, int j) {
        // g tile: rows j..j+32, 4 KB contiguous; 2 x 16B per thread
        unsigned gl = (unsigned)(uintptr_t)(&lds_g[buf][t * 16]);
        unsigned go = (unsigned)((j * LAT + t * 16) * (int)sizeof(bf16_t));
        async_ld16(gl,      g, go);
        async_ld16(gl + 16, g, go + 16);
        // hT tile: thread t owns feature row n=t, copies hT[n][j..j+32] (64 B)
        unsigned hl = (unsigned)(uintptr_t)(&lds_hT[buf][t * 32]);
        unsigned ho = (unsigned)(((unsigned)t * NN + (unsigned)j) * (unsigned)sizeof(bf16_t));
        #pragma unroll
        for (int b2 = 0; b2 < 4; ++b2)
            async_ld16(hl + b2 * 16, hT, ho + b2 * 16);
    };

    // B-operand fetch helpers (contiguous k order, same as the Ag A operands)
    auto loadBg = [&](const bf16_t* Lg, int s, int ks) -> V16 {
        V16 b;   // lane: col n = l15 (tile s), K = ks*32 + half*16 + 0..15
        const bf16_t* src = Lg + (s * 16 + l15) * LAT + ks * 32 + half * 16;
        b.q[0] = *(const u32x4*)(src);
        b.q[1] = *(const u32x4*)(src + 8);
        return b;
    };
    auto loadBh = [&](const bf16_t* Lh, int n) -> V16 {
        V16 b;   // lane: col n*16+l15, K = half*16 + 0..15 from hT tile
        const bf16_t* src = Lh + (n * 16 + l15) * 32 + half * 16;
        b.q[0] = *(const u32x4*)(src);
        b.q[1] = *(const u32x4*)(src + 8);
        return b;
    };

    issue_tile(0, jb);
    wait_async0();
    __syncthreads();

    int cur = 0;
    for (int jj = jb; jj < je; jj += 32) {
        if (jj + 32 < je) issue_tile(cur ^ 1, jj + 32);   // overlap DMA w/ math

        const bf16_t* Lg = lds_g[cur];
        const bf16_t* Lh = lds_hT[cur];

        // --- S = g_i @ g_j^T : two 16x16 f32 tiles; ks-outer so consecutive
        //     WMMAs share the A operand -> reuse_a on the second of each pair.
        f32x8 S[2] = {zero8, zero8};
        #pragma unroll
        for (int ks = 0; ks < 2; ++ks) {
            V16 b0 = loadBg(Lg, 0, ks);
            V16 b1 = loadBg(Lg, 1, ks);
            S[0] = wmma_bf16<false>(Ag[ks], b0.v, S[0]);
            S[1] = wmma_bf16<true >(Ag[ks], b1.v, S[1]);
        }

        // --- adj = sigmoid(2t*S - t*(sq_i+sq_j) + th); stage P in LDS -------
        const float sj0 = sq[jj + l15];
        const float sj1 = sq[jj + 16 + l15];
        bf16_t* P = lds_P[wv];
        #pragma unroll
        for (int v = 0; v < 8; ++v) {
            const float z0 = 2.0f * tt * S[0][v] - tt * (sqi[v] + sj0) + th;
            const float z1 = 2.0f * tt * S[1][v] - tt * (sqi[v] + sj1) + th;
            const float a0 = __builtin_amdgcn_rcpf(1.0f + __expf(-z0));
            const float a1 = __builtin_amdgcn_rcpf(1.0f + __expf(-z1));
            dpart[v] += a0 + a1;
            const int M = v + half * 8;     // C/D layout row
            P[M * 32 + l15]      = (bf16_t)a0;
            P[M * 32 + 16 + l15] = (bf16_t)a1;
        }
        asm volatile("s_wait_dscnt 0" ::: "memory");

        // --- re-load P in A layout (16x32); acc += P @ h_j.
        //     All 8 WMMAs share A -> reuse_a on 2..8 (peeled first iter).
        V16 ap;
        {
            const bf16_t* src = P + l15 * 32 + half * 16;
            ap.q[0] = *(const u32x4*)(src);
            ap.q[1] = *(const u32x4*)(src + 8);
        }
        {
            V16 b = loadBh(Lh, 0);
            acc[0] = wmma_bf16<false>(ap.v, b.v, acc[0]);
        }
        #pragma unroll
        for (int n = 1; n < 8; ++n) {
            V16 b = loadBh(Lh, n);
            acc[n] = wmma_bf16<true>(ap.v, b.v, acc[n]);
        }

        wait_async0();       // my next-tile DMA done
        __syncthreads();     // everyone's DMA visible, all reads of nxt done
        cur ^= 1;
    }

    // --- reduce row-sums across the 16 lanes of each half; write partials ---
    #pragma unroll
    for (int v = 0; v < 8; ++v) {
        float d = dpart[v];
        d += __shfl_xor(d, 8, 16);
        d += __shfl_xor(d, 4, 16);
        d += __shfl_xor(d, 2, 16);
        d += __shfl_xor(d, 1, 16);
        if (l15 == 0)
            pden[(size_t)cs * NN + ibase + v + half * 8] = d;
    }
    #pragma unroll
    for (int n = 0; n < 8; ++n) {
        #pragma unroll
        for (int v = 0; v < 8; ++v) {
            const int M = v + half * 8;
            pacc[((size_t)cs * NN + ibase + M) * HID + n * 16 + l15] = acc[n][v];
        }
    }
}

// ---------------------------------------------------------------------------
// Combine the CSPLIT partials: out[i,n] = (sum_cs pacc) / (sum_cs pden), +relu
// ---------------------------------------------------------------------------
__global__ __launch_bounds__(256) void normalize_kernel(
    const float* __restrict__ pacc, const float* __restrict__ pden,
    float* __restrict__ out, const int apply_relu)
{
    const size_t idx = (size_t)blockIdx.x * 256 + threadIdx.x;  // over N*HID
    const size_t row = idx >> 7;
    float a = 0.0f, d = 0.0f;
    #pragma unroll
    for (int cs = 0; cs < CSPLIT; ++cs) {
        a += pacc[((size_t)cs * NN) * HID + idx];
        d += pden[(size_t)cs * NN + row];
    }
    float val = a * __builtin_amdgcn_rcpf(d);
    if (apply_relu) val = fmaxf(val, 0.0f);
    out[idx] = val;
}

// ---------------------------------------------------------------------------
// Output head: logits = x@out_w + out_b ; softmax. One thread per row.
// ---------------------------------------------------------------------------
__global__ __launch_bounds__(256) void head_kernel(
    const float* __restrict__ x, const float* __restrict__ w,
    const float* __restrict__ b, float* __restrict__ out)
{
    const int r = blockIdx.x * blockDim.x + threadIdx.x;
    if (r >= NN) return;
    float lg[10];
    #pragma unroll
    for (int j = 0; j < 10; ++j) lg[j] = b[j];
    for (int k = 0; k < DIN; ++k) {
        const float xv = x[(size_t)r * DIN + k];
        #pragma unroll
        for (int j = 0; j < 10; ++j) lg[j] = fmaf(xv, w[k * 10 + j], lg[j]);
    }
    float mx = lg[0];
    #pragma unroll
    for (int j = 1; j < 10; ++j) mx = fmaxf(mx, lg[j]);
    float s = 0.0f;
    #pragma unroll
    for (int j = 0; j < 10; ++j) { lg[j] = __expf(lg[j] - mx); s += lg[j]; }
    const float inv = __builtin_amdgcn_rcpf(s);
    #pragma unroll
    for (int j = 0; j < 10; ++j) out[(size_t)r * 10 + j] = lg[j] * inv;
}

// ---------------------------------------------------------------------------
extern "C" void kernel_launch(void* const* d_in, const int* in_sizes, int n_in,
                              void* d_out, int out_size, void* d_ws, size_t ws_size,
                              hipStream_t stream)
{
    const float* feat   = (const float*)d_in[0];
    const float* gl_w0  = (const float*)d_in[1];
    const float* gl_b0  = (const float*)d_in[2];
    const float* gl_w1  = (const float*)d_in[3];
    const float* gl_b1  = (const float*)d_in[4];
    const float* gnn_w0 = (const float*)d_in[5];
    const float* gnn_b0 = (const float*)d_in[6];
    const float* gnn_w1 = (const float*)d_in[7];
    const float* gnn_b1 = (const float*)d_in[8];
    const float* out_w  = (const float*)d_in[9];
    const float* out_b  = (const float*)d_in[10];
    const float* temp   = (const float*)d_in[11];
    const float* theta  = (const float*)d_in[12];
    float* outp = (float*)d_out;

    char* ws = (char*)d_ws;
    bf16_t* g_bf  = (bf16_t*)(ws);                            // [0,2) MB
    bf16_t* hT_bf = (bf16_t*)(ws + ((size_t)2 << 20));        // [2,6) MB
    float*  sqv   = (float*) (ws + ((size_t)6 << 20));        // 64 KB
    float*  xbuf  = (float*) (ws + ((size_t)8 << 20));        // [8,16) MB
    float*  pacc  = (float*) (ws + ((size_t)16 << 20));       // [16,48) MB
    float*  pden  = (float*) (ws + ((size_t)48 << 20));       // 256 KB

    // layer 0
    proj_kernel<<<NN, 128, 0, stream>>>(feat, gl_w0, gl_b0, gnn_w0, gnn_b0,
                                        g_bf, sqv, hT_bf);
    fused_kernel<<<256 * CSPLIT, 128, 0, stream>>>(g_bf, sqv, hT_bf, temp, theta,
                                                   pacc, pden);
    normalize_kernel<<<NN * HID / 256, 256, 0, stream>>>(pacc, pden, xbuf, 1);
    // layer 1 (no relu on output)
    proj_kernel<<<NN, 128, 0, stream>>>(xbuf, gl_w1, gl_b1, gnn_w1, gnn_b1,
                                        g_bf, sqv, hT_bf);
    fused_kernel<<<256 * CSPLIT, 128, 0, stream>>>(g_bf, sqv, hT_bf, temp, theta,
                                                   pacc, pden);
    normalize_kernel<<<NN * HID / 256, 256, 0, stream>>>(pacc, pden, xbuf, 0);
    // head + softmax
    head_kernel<<<NN / 256, 256, 0, stream>>>(xbuf, out_w, out_b, outp);
}
